// LSTMRecursive_13383118094563
// MI455X (gfx1250) — matrix operand order repaired and checked by reading, not compiled, not run on hardware
//
#include <hip/hip_runtime.h>

#define B_       4096
#define SEQ      169
#define T_       168
#define H_       64
#define G_       256     // 4*H gates
#define NSTEPS   24
#define BM       16      // batch rows per workgroup
#define NTHREADS 256     // 8 waves

typedef _Float16 v16h __attribute__((ext_vector_type(16)));
typedef _Float16 v4h  __attribute__((ext_vector_type(4)));
typedef float    v8f  __attribute__((ext_vector_type(8)));

__device__ __forceinline__ float fast_tanh(float x) {
#if __has_builtin(__builtin_amdgcn_tanhf)
  return __builtin_amdgcn_tanhf(x);           // v_tanh_f32 (gfx1250 trans op)
#else
  float e = __expf(2.0f * x);
  return 1.0f - 2.0f / (e + 1.0f);
#endif
}
__device__ __forceinline__ float fast_sig(float x) {
  return 0.5f * fast_tanh(0.5f * x) + 0.5f;   // 1 trans op per sigmoid
}

__global__ __launch_bounds__(NTHREADS) void lstm_recursive_kernel(
    const float* __restrict__ x,
    const float* __restrict__ Wih1, const float* __restrict__ Whh1,
    const float* __restrict__ bih1, const float* __restrict__ bhh1,
    const float* __restrict__ Wih2, const float* __restrict__ Whh2,
    const float* __restrict__ bih2, const float* __restrict__ bhh2,
    const float* __restrict__ fc1w, const float* __restrict__ fc1b,
    const float* __restrict__ fc2w, const float* __restrict__ fc2b,
    float* __restrict__ out, _Float16* __restrict__ ws)
{
  __shared__ float     slab[BM * G_];     // 16 KB gate pre-activations
  __shared__ _Float16  hfrag[1024];       // 2 KB: h in WMMA A-fragment layout (2 K-frags)
  __shared__ float     winbuf[BM][T_];    // circular input window
  __shared__ float     weff[H_ + 1];      // fused fc weights
  __shared__ float     dbuf[BM];
  __shared__ float     beff;

  const int tid  = threadIdx.x;
  const int wave = tid >> 5;
  const int lane = tid & 31;
  const int nl   = lane & 15;
  const int lgrp = lane >> 4;
  const int wg   = blockIdx.x;
  const int rowbase = wg * BM;

  // ---- fuse fc1/fc2 (no nonlinearity between them): out = feat . weff + beff ----
  if (tid < H_ + 1) {
    float a = 0.f;
    for (int j = 0; j < 64; ++j) a += fc2w[j] * fc1w[j * 65 + tid];
    weff[tid] = a;
  } else if (tid == H_ + 1) {
    float a = fc2b[0];
    for (int j = 0; j < 64; ++j) a += fc2w[j] * fc1b[j];
    beff = a;
  }

  // ---- input window + last sample ----
  for (int i = tid; i < BM * T_; i += NTHREADS) {
    int m = i / T_, t = i % T_;
    winbuf[m][t] = x[(size_t)(rowbase + m) * SEQ + t];
  }
  if (tid < BM) dbuf[tid] = x[(size_t)(rowbase + tid) * SEQ + T_];
  for (int i = tid; i < 1024; i += NTHREADS) hfrag[i] = (_Float16)0.f;

  // ---- per-wave resident weight B-fragments (B[k][n]: lane = n + 16*(k/16), elem = k%16) ----
  v16h wh1[2][2], wh2[2][2], wx2[2][2];
  #pragma unroll
  for (int nt = 0; nt < 2; ++nt) {
    const int gcol = wave * 32 + nt * 16 + nl;
    #pragma unroll
    for (int kf = 0; kf < 2; ++kf) {
      const int kb = kf * 32 + lgrp * 16;
      #pragma unroll
      for (int e = 0; e < 16; ++e) {
        wh1[nt][kf][e] = (_Float16)Whh1[gcol * H_ + kb + e];
        wh2[nt][kf][e] = (_Float16)Whh2[gcol * H_ + kb + e];
        wx2[nt][kf][e] = (_Float16)Wih2[gcol * H_ + kb + e];
      }
    }
  }
  // column-only biases folded into accumulator init
  const int col0 = wave * 32 + nl;
  const float bias1_0 = bih1[col0] + bhh1[col0];
  const float bias1_1 = bih1[col0 + 16] + bhh1[col0 + 16];
  const float bias2_0 = bih2[col0] + bhh2[col0];
  const float bias2_1 = bih2[col0 + 16] + bhh2[col0 + 16];

  // ---- elementwise mapping: thread -> (row m, 4 hidden cols j0..j0+3) ----
  const int m  = tid >> 4;
  const int j0 = (tid & 15) * 4;
  float w1v[4][4];                              // Wih1 column values (input dim = 1)
  #pragma unroll
  for (int g = 0; g < 4; ++g)
    #pragma unroll
    for (int q = 0; q < 4; ++q) w1v[g][q] = Wih1[g * H_ + j0 + q];

  // A-fragment slot for this thread's 4 h values (16-bit A 16x32 layout)
  const int kk    = j0 & 31;
  const int frag  = j0 >> 5;
  const int wl    = m + 16 * (kk >> 4);
  const int e0    = kk & 15;
  const int hslot = frag * 512 + wl * 16 + e0;

  _Float16* __restrict__ h1s = ws + (size_t)wg * T_ * 1024;   // relu(h1) stream, frag layout

  float c[4];
  __syncthreads();

  for (int s = 0; s < NSTEPS; ++s) {
    // ================= LSTM layer 1 (input dim 1) =================
    #pragma unroll
    for (int q = 0; q < 4; ++q) c[q] = 0.f;
    for (int t = 0; t < T_; ++t) {
      v16h a0 = *(const v16h*)&hfrag[lane * 16];
      v16h a1 = *(const v16h*)&hfrag[512 + lane * 16];
      v8f acc0, acc1;
      #pragma unroll
      for (int r = 0; r < 8; ++r) { acc0[r] = bias1_0; acc1[r] = bias1_1; }
      acc0 = __builtin_amdgcn_wmma_f32_16x16x32_f16(false, a0, false, wh1[0][0], (short)0, acc0, false, false);
      acc0 = __builtin_amdgcn_wmma_f32_16x16x32_f16(false, a1, false, wh1[0][1], (short)0, acc0, false, false);
      acc1 = __builtin_amdgcn_wmma_f32_16x16x32_f16(false, a0, false, wh1[1][0], (short)0, acc1, false, false);
      acc1 = __builtin_amdgcn_wmma_f32_16x16x32_f16(false, a1, false, wh1[1][1], (short)0, acc1, false, false);
      #pragma unroll
      for (int r = 0; r < 8; ++r) {
        const int row = r + 8 * lgrp;
        slab[row * G_ + col0]      = acc0[r];
        slab[row * G_ + col0 + 16] = acc1[r];
      }
      __syncthreads();
      // elementwise cell update
      const float xt = winbuf[m][(s + t) % T_];
      const float* gp = &slab[m * G_ + j0];
      float hv[4];
      #pragma unroll
      for (int q = 0; q < 4; ++q) {
        float ii  = gp[q]       + xt * w1v[0][q];
        float ff  = gp[64 + q]  + xt * w1v[1][q];
        float ggv = gp[128 + q] + xt * w1v[2][q];
        float oo  = gp[192 + q] + xt * w1v[3][q];
        c[q]  = fast_sig(ff) * c[q] + fast_sig(ii) * fast_tanh(ggv);
        hv[q] = fast_sig(oo) * fast_tanh(c[q]);
      }
      v4h hp, hr;
      #pragma unroll
      for (int q = 0; q < 4; ++q) {
        hp[q] = (_Float16)hv[q];
        hr[q] = (_Float16)fmaxf(hv[q], 0.f);
      }
      *(v4h*)&hfrag[hslot] = hp;                        // recurrent h (no relu)
      *(v4h*)&h1s[(size_t)t * 1024 + hslot] = hr;       // relu(h1) for layer 2
      __syncthreads();
    }
    // at this point: hfrag = h1_T, c[] = c1_T  -> seed layer 2

    // ================= LSTM layer 2 (input = relu(h1s)) =================
    for (int t = 0; t < T_; ++t) {
      const _Float16* xa = &h1s[(size_t)t * 1024];
      v16h ax0 = *(const v16h*)&xa[lane * 16];
      v16h ax1 = *(const v16h*)&xa[512 + lane * 16];
      v16h a0  = *(const v16h*)&hfrag[lane * 16];
      v16h a1  = *(const v16h*)&hfrag[512 + lane * 16];
      v8f acc0, acc1;
      #pragma unroll
      for (int r = 0; r < 8; ++r) { acc0[r] = bias2_0; acc1[r] = bias2_1; }
      acc0 = __builtin_amdgcn_wmma_f32_16x16x32_f16(false, ax0, false, wx2[0][0], (short)0, acc0, false, false);
      acc0 = __builtin_amdgcn_wmma_f32_16x16x32_f16(false, ax1, false, wx2[0][1], (short)0, acc0, false, false);
      acc0 = __builtin_amdgcn_wmma_f32_16x16x32_f16(false, a0,  false, wh2[0][0], (short)0, acc0, false, false);
      acc0 = __builtin_amdgcn_wmma_f32_16x16x32_f16(false, a1,  false, wh2[0][1], (short)0, acc0, false, false);
      acc1 = __builtin_amdgcn_wmma_f32_16x16x32_f16(false, ax0, false, wx2[1][0], (short)0, acc1, false, false);
      acc1 = __builtin_amdgcn_wmma_f32_16x16x32_f16(false, ax1, false, wx2[1][1], (short)0, acc1, false, false);
      acc1 = __builtin_amdgcn_wmma_f32_16x16x32_f16(false, a0,  false, wh2[1][0], (short)0, acc1, false, false);
      acc1 = __builtin_amdgcn_wmma_f32_16x16x32_f16(false, a1,  false, wh2[1][1], (short)0, acc1, false, false);
      #pragma unroll
      for (int r = 0; r < 8; ++r) {
        const int row = r + 8 * lgrp;
        slab[row * G_ + col0]      = acc0[r];
        slab[row * G_ + col0 + 16] = acc1[r];
      }
      __syncthreads();
      const float* gp = &slab[m * G_ + j0];
      float hv[4];
      #pragma unroll
      for (int q = 0; q < 4; ++q) {
        float ii  = gp[q];
        float ff  = gp[64 + q];
        float ggv = gp[128 + q];
        float oo  = gp[192 + q];
        c[q]  = fast_sig(ff) * c[q] + fast_sig(ii) * fast_tanh(ggv);
        hv[q] = fast_sig(oo) * fast_tanh(c[q]);
      }
      if (t == T_ - 1) {
        // fused FC head on relu(h2_T) + window append
        float partial = 0.f;
        #pragma unroll
        for (int q = 0; q < 4; ++q) partial += fmaxf(hv[q], 0.f) * weff[j0 + q];
        for (int off = 8; off >= 1; off >>= 1)
          partial += __shfl_xor(partial, off, 16);
        if ((tid & 15) == 0) {
          float o = partial + weff[H_] * dbuf[m] + beff;
          out[(size_t)(rowbase + m) * NSTEPS + s] = o;
          winbuf[m][s % T_] = o;                        // circular append
        }
        v4h z;
        #pragma unroll
        for (int q = 0; q < 4; ++q) z[q] = (_Float16)0.f;
        *(v4h*)&hfrag[hslot] = z;                       // reset h for next outer step
      } else {
        v4h hp;
        #pragma unroll
        for (int q = 0; q < 4; ++q) hp[q] = (_Float16)hv[q];
        *(v4h*)&hfrag[hslot] = hp;
      }
      __syncthreads();
    }
  }
}

extern "C" void kernel_launch(void* const* d_in, const int* in_sizes, int n_in,
                              void* d_out, int out_size, void* d_ws, size_t ws_size,
                              hipStream_t stream) {
  (void)in_sizes; (void)n_in; (void)out_size; (void)ws_size;
  const float* x    = (const float*)d_in[0];
  const float* Wih1 = (const float*)d_in[1];
  const float* Whh1 = (const float*)d_in[2];
  const float* bih1 = (const float*)d_in[3];
  const float* bhh1 = (const float*)d_in[4];
  const float* Wih2 = (const float*)d_in[5];
  const float* Whh2 = (const float*)d_in[6];
  const float* bih2 = (const float*)d_in[7];
  const float* bhh2 = (const float*)d_in[8];
  const float* fc1w = (const float*)d_in[9];
  const float* fc1b = (const float*)d_in[10];
  const float* fc2w = (const float*)d_in[11];
  const float* fc2b = (const float*)d_in[12];
  lstm_recursive_kernel<<<B_ / BM, NTHREADS, 0, stream>>>(
      x, Wih1, Whh1, bih1, bhh1, Wih2, Whh2, bih2, bhh2,
      fc1w, fc1b, fc2w, fc2b, (float*)d_out, (_Float16*)d_ws);
}
